// BiMambaBlock_25005299597897
// MI455X (gfx1250) — matrix operand order repaired and checked by reading, not compiled, not run on hardware
//
#include <hip/hip_runtime.h>

#define BATCHN 2
#define SEQL   1024
#define DM     1024
#define DI     2048
#define DRANK  64
#define DSTATE 16
#define XDIM   96            // DRANK + 2*DSTATE
#define BL     (BATCHN * SEQL)   // 2048

typedef __attribute__((ext_vector_type(16))) __bf16        v16bf;
typedef __attribute__((ext_vector_type(8)))  float         v8f;
typedef __attribute__((ext_vector_type(4)))  unsigned int  u32x4;

union FragBF {
    v16bf v;
    u32x4 q[2];
};

#if defined(__has_builtin)
#if __has_builtin(__builtin_amdgcn_global_load_async_to_lds_b32)
#define USE_ASYNC_LDS 1
#endif
#endif
#ifndef USE_ASYNC_LDS
#define USE_ASYNC_LDS 0
#endif

#if USE_ASYNC_LDS
typedef __attribute__((address_space(1))) int* as1_int_p;   // global
typedef __attribute__((address_space(3))) int* as3_int_p;   // LDS
#endif

// ---------------------------------------------------------------------------
// LayerNorm + residual init + bf16 cast + reversed copy
// one block per row (B*L rows), 256 threads
// ---------------------------------------------------------------------------
__global__ __launch_bounds__(256)
void ln_residual_kernel(const float* __restrict__ x, const float* __restrict__ gamma,
                        const float* __restrict__ beta, __bf16* __restrict__ xn,
                        __bf16* __restrict__ xn_rev, float* __restrict__ out)
{
    const int row = blockIdx.x;
    const int b = row / SEQL, l = row % SEQL;
    const float* xr = x + (size_t)row * DM;
    __shared__ float red[256];

    float s = 0.f;
    for (int i = threadIdx.x; i < DM; i += 256) s += xr[i];
    red[threadIdx.x] = s; __syncthreads();
    for (int st = 128; st > 0; st >>= 1) {
        if (threadIdx.x < st) red[threadIdx.x] += red[threadIdx.x + st];
        __syncthreads();
    }
    const float mu = red[0] * (1.f / DM);
    __syncthreads();

    float vv = 0.f;
    for (int i = threadIdx.x; i < DM; i += 256) { float d = xr[i] - mu; vv += d * d; }
    red[threadIdx.x] = vv; __syncthreads();
    for (int st = 128; st > 0; st >>= 1) {
        if (threadIdx.x < st) red[threadIdx.x] += red[threadIdx.x + st];
        __syncthreads();
    }
    const float rstd = rsqrtf(red[0] * (1.f / DM) + 1e-5f);

    const size_t rrow = (size_t)(b * SEQL + (SEQL - 1 - l)) * DM;
    for (int i = threadIdx.x; i < DM; i += 256) {
        float xv = xr[i];
        out[(size_t)row * DM + i] = xv;                       // residual
        float nv = (xv - mu) * rstd * gamma[i] + beta[i];
        __bf16 h = (__bf16)nv;
        xn[(size_t)row * DM + i] = h;
        xn_rev[rrow + i] = h;
    }
}

// ---------------------------------------------------------------------------
// f32 -> bf16 cast
// ---------------------------------------------------------------------------
__global__ void cvt_bf16_kernel(const float* __restrict__ s, __bf16* __restrict__ d, int n)
{
    int i = blockIdx.x * 256 + threadIdx.x;
    if (i < n) d[i] = (__bf16)s[i];
}

// ---------------------------------------------------------------------------
// WMMA bf16 GEMM:  C[M,N] (f32) = A[M,K] (bf16, row-major) * W[N,K]^T (bf16)
// block = 256 threads = 8 waves; block tile 128x128; wave tile 64x32.
// Software-pipelined K loop (two static fragment buffers, K % 64 == 0).
// B-edge handled by clamping the loaded row (stores are guarded), so the
// hot loop carries no EXEC-mask manipulation (WMMA requires EXEC all-ones).
// EPI 0: store  |  EPI 1: softplus(acc + bias[n])  |  EPI 2: C[m'] += acc
//   (EPI 2: m' row-reversed per batch when rev_L != 0)
// Requires M % 128 == 0, K % 64 == 0 (true for all call sites); N guarded.
// ---------------------------------------------------------------------------
#define LOAD_STAGE(buf, koff)                                                  \
    do {                                                                       \
        _Pragma("unroll") for (int tm = 0; tm < 4; ++tm) {                     \
            const __bf16* p = aptr[tm] + (koff);                               \
            a[buf][tm].q[0] = *(const u32x4*)p;                                \
            a[buf][tm].q[1] = *(const u32x4*)(p + 16);                         \
        }                                                                      \
        _Pragma("unroll") for (int tn = 0; tn < 2; ++tn) {                     \
            const __bf16* p = bptr[tn] + (koff);                               \
            bb[buf][tn].q[0] = *(const u32x4*)p;                               \
            bb[buf][tn].q[1] = *(const u32x4*)(p + 16);                        \
        }                                                                      \
    } while (0)

#define WMMA_STAGE(buf)                                                        \
    do {                                                                       \
        _Pragma("unroll") for (int tm = 0; tm < 4; ++tm)                       \
        _Pragma("unroll") for (int tn = 0; tn < 2; ++tn)                       \
            acc[tm][tn] = __builtin_amdgcn_wmma_f32_16x16x32_bf16(             \
                false, a[buf][tm].v, false, bb[buf][tn].v, (short)0,           \
                acc[tm][tn], false, false);                                    \
    } while (0)

template<int EPI>
__global__ __launch_bounds__(256)
void gemm_wmma_bf16(const __bf16* __restrict__ A, const __bf16* __restrict__ W,
                    float* __restrict__ C, int M, int N, int K,
                    const float* __restrict__ bias, int rev_L)
{
    const int lane = threadIdx.x & 31;
    const int wave = threadIdx.x >> 5;
    const int wm   = wave >> 2;          // 0..1
    const int wn   = wave & 3;           // 0..3
    const int bm   = blockIdx.x * 128;
    const int bn   = blockIdx.y * 128;
    const int r    = lane & 15;
    const int grp  = lane >> 4;

    v8f acc[4][2];
#pragma unroll
    for (int i = 0; i < 4; ++i)
#pragma unroll
        for (int j = 0; j < 2; ++j)
            acc[i][j] = (v8f){0.f,0.f,0.f,0.f,0.f,0.f,0.f,0.f};

    const int arow0 = bm + wm * 64 + r;      // A row for this lane (per tm tile +16)
    const int bcol0 = bn + wn * 32 + r;      // W row (= B column) for this lane

    // Per-lane base pointers.
    // A-frag: lane holds K in {grp*8 .. grp*8+7} and {grp*8+16 .. grp*8+23}
    // B-frag: lane holds its column's K in {grp*8 .. grp*8+7} and {grp*8+16 .. grp*8+23}
    const __bf16* aptr[4];
#pragma unroll
    for (int tm = 0; tm < 4; ++tm)
        aptr[tm] = A + (size_t)(arow0 + tm * 16) * K + grp * 8;
    const __bf16* bptr[2];
#pragma unroll
    for (int tn = 0; tn < 2; ++tn) {
        int n = bcol0 + tn * 16;
        if (n > N - 1) n = N - 1;            // clamp: garbage cols, stores guarded
        bptr[tn] = W + (size_t)n * K + grp * 8;
    }

    FragBF a[2][4], bb[2][2];

    LOAD_STAGE(0, 0);
    for (int k0 = 0; k0 < K; k0 += 64) {
        LOAD_STAGE(1, k0 + 32);
        if (k0 + 64 < K) {
            __builtin_prefetch(aptr[0] + k0 + 64, 0, 0);
            __builtin_prefetch(bptr[0] + k0 + 64, 0, 0);
        }
        WMMA_STAGE(0);
        if (k0 + 64 < K) LOAD_STAGE(0, k0 + 64);
        WMMA_STAGE(1);
    }

    // C/D layout: lane -> n = lane&15, m = (lane>>4)*8 + vgpr_index
    const int mlo = grp * 8;
#pragma unroll
    for (int tm = 0; tm < 4; ++tm) {
#pragma unroll
        for (int tn = 0; tn < 2; ++tn) {
            const int n = bn + wn * 32 + tn * 16 + r;
            if (n >= N) continue;
#pragma unroll
            for (int rr = 0; rr < 8; ++rr) {
                const int m = bm + wm * 64 + tm * 16 + mlo + rr;
                if (m >= M) continue;
                float v = acc[tm][tn][rr];
                if (EPI == 1) {
                    float t = v + bias[n];
                    v = (t > 20.f) ? t : log1pf(__expf(t));   // softplus
                }
                if (EPI == 2) {
                    int mo = m;
                    if (rev_L) {
                        int bbx = m / rev_L;
                        int ll  = m - bbx * rev_L;
                        mo = bbx * rev_L + (rev_L - 1 - ll);
                    }
                    C[(size_t)mo * N + n] += v;
                } else {
                    C[(size_t)m * N + n] = v;
                }
            }
        }
    }
}

#undef LOAD_STAGE
#undef WMMA_STAGE

// ---------------------------------------------------------------------------
// depthwise causal conv (width 4) + bias + SiLU over xz[..., :DI]
// ---------------------------------------------------------------------------
__global__ __launch_bounds__(256)
void conv_silu_kernel(const float* __restrict__ xz, const float* __restrict__ cw,
                      const float* __restrict__ cb, float* __restrict__ u,
                      __bf16* __restrict__ uh)
{
    const int idx = blockIdx.x * 256 + threadIdx.x;
    if (idx >= BL * DI) return;
    const int e  = idx & (DI - 1);
    const int bl = idx / DI;
    const int l  = bl & (SEQL - 1);
    float acc = cb[e];
#pragma unroll
    for (int j = 0; j < 4; ++j) {
        int ls = l - 3 + j;
        if (ls >= 0) acc += cw[e * 4 + j] * xz[(size_t)(bl - 3 + j) * (2 * DI) + e];
    }
    float sv = acc / (1.f + __expf(-acc));    // SiLU
    u[idx]  = sv;
    uh[idx] = (__bf16)sv;
}

// ---------------------------------------------------------------------------
// copy first DRANK cols of x_dbl (stride XDIM) into dense bf16 (stride DRANK)
// ---------------------------------------------------------------------------
__global__ void slice_cast_kernel(const float* __restrict__ xd, __bf16* __restrict__ o)
{
    int i = blockIdx.x * 256 + threadIdx.x;
    if (i >= BL * DRANK) return;
    int m = i >> 6, rr = i & 63;
    o[i] = (__bf16)xd[m * XDIM + rr];
}

// ---------------------------------------------------------------------------
// selective scan: single-wave (32-thread) blocks -> barrier ops degenerate to
// S_NOP; one thread per (b, channel e); 16-wide state in registers.
// B/C broadcast per timestep staged via gfx1250 async global->LDS copy
// (ASYNCcnt + s_wait_asynccnt) when available; fused +u*D and SiLU(z) gate.
// grid = BATCHN * (DI/32) = 128 blocks of 32
// ---------------------------------------------------------------------------
__global__ __launch_bounds__(32)
void scan_kernel(const float* __restrict__ dt, const float* __restrict__ u,
                 const float* __restrict__ xz, const float* __restrict__ xd,
                 const float* __restrict__ A_log, const float* __restrict__ Dp,
                 __bf16* __restrict__ yg)
{
    const int b = blockIdx.x >> 6;                        // DI/32 == 64
    const int e = ((blockIdx.x & 63) << 5) + threadIdx.x;
    __shared__ float BC[2 * DSTATE];                      // B | C

    float a[DSTATE], h[DSTATE];
#pragma unroll
    for (int n = 0; n < DSTATE; ++n) {
        a[n] = -__expf(A_log[e * DSTATE + n]);
        h[n] = 0.f;
    }
    const float dval = Dp[e];

    for (int l = 0; l < SEQL; ++l) {
        const size_t row = (size_t)b * SEQL + l;
        // stage B (16) and C (16) for this timestep: one f32 per lane
        {
            const float* g = xd + row * XDIM + DRANK + threadIdx.x;
#if USE_ASYNC_LDS
            __builtin_amdgcn_global_load_async_to_lds_b32(
                (as1_int_p)(void*)g,
                (as3_int_p)(void*)&BC[threadIdx.x],
                0, 0);
            asm volatile("s_wait_asynccnt 0x0" ::: "memory");
#else
            BC[threadIdx.x] = *g;
#endif
        }
        __syncthreads();

        const float dtv = dt[row * DI + e];
        const float uv  = u[row * DI + e];
        float y = 0.f;
#pragma unroll
        for (int n = 0; n < DSTATE; ++n) {
            h[n] = __expf(dtv * a[n]) * h[n] + dtv * BC[n] * uv;
            y += h[n] * BC[DSTATE + n];
        }
        y += uv * dval;
        const float zv = xz[row * (2 * DI) + DI + e];
        y *= zv / (1.f + __expf(-zv));                  // * SiLU(z)
        yg[row * DI + e] = (__bf16)y;
        __syncthreads();
    }
}

// ---------------------------------------------------------------------------
// host launcher
// ---------------------------------------------------------------------------
extern "C" void kernel_launch(void* const* d_in, const int* in_sizes, int n_in,
                              void* d_out, int out_size, void* d_ws, size_t ws_size,
                              hipStream_t stream)
{
    const float* x     = (const float*)d_in[0];
    const float* gamma = (const float*)d_in[1];
    const float* beta  = (const float*)d_in[2];

    struct Dir { const float *in_w, *conv_w, *conv_b, *xproj_w, *dt_w, *dt_b, *A_log, *D, *out_w; };
    Dir dirs[2];
    for (int d = 0; d < 2; ++d) {
        const int base = 3 + d * 9;
        dirs[d].in_w    = (const float*)d_in[base + 0];
        dirs[d].conv_w  = (const float*)d_in[base + 1];
        dirs[d].conv_b  = (const float*)d_in[base + 2];
        dirs[d].xproj_w = (const float*)d_in[base + 3];
        dirs[d].dt_w    = (const float*)d_in[base + 4];
        dirs[d].dt_b    = (const float*)d_in[base + 5];
        dirs[d].A_log   = (const float*)d_in[base + 6];
        dirs[d].D       = (const float*)d_in[base + 7];
        dirs[d].out_w   = (const float*)d_in[base + 8];
    }

    char* ws = (char*)d_ws;
    size_t off = 0;
    auto take = [&](size_t bytes) {
        char* p = ws + off;
        off += (bytes + 255) & ~(size_t)255;
        return (void*)p;
    };
    __bf16* xn     = (__bf16*)take((size_t)BL * DM * 2);
    __bf16* xn_rev = (__bf16*)take((size_t)BL * DM * 2);
    __bf16* wbuf   = (__bf16*)take((size_t)(2 * DI) * DM * 2);  // weight scratch (max 8 MB)
    float*  xz     = (float*) take((size_t)BL * (2 * DI) * 4);
    float*  ubuf   = (float*) take((size_t)BL * DI * 4);
    __bf16* uh     = (__bf16*)take((size_t)BL * DI * 2);
    float*  xdbl   = (float*) take((size_t)BL * XDIM * 4);
    __bf16* dtin   = (__bf16*)take((size_t)BL * DRANK * 2);
    float*  dtbuf  = (float*) take((size_t)BL * DI * 4);
    __bf16* ybuf   = (__bf16*)take((size_t)BL * DI * 2);
    (void)ws_size; (void)in_sizes; (void)n_in; (void)out_size;

    float* out = (float*)d_out;

    // LayerNorm + residual init + fwd/rev bf16 activations
    ln_residual_kernel<<<BL, 256, 0, stream>>>(x, gamma, beta, xn, xn_rev, out);

    for (int d = 0; d < 2; ++d) {
        const Dir& P = dirs[d];
        const __bf16* xin = d ? xn_rev : xn;

        // in-proj: xz = xin @ in_w^T    (2048 x 4096, K=1024)
        {
            const int n = (2 * DI) * DM;
            cvt_bf16_kernel<<<(n + 255) / 256, 256, 0, stream>>>(P.in_w, wbuf, n);
            dim3 g(BL / 128, (2 * DI) / 128);
            gemm_wmma_bf16<0><<<g, 256, 0, stream>>>(xin, wbuf, xz, BL, 2 * DI, DM, nullptr, 0);
        }

        // depthwise causal conv + SiLU on u half
        conv_silu_kernel<<<(BL * DI) / 256, 256, 0, stream>>>(xz, P.conv_w, P.conv_b, ubuf, uh);

        // x-proj: xdbl = u @ xproj_w^T  (2048 x 96, K=2048)
        {
            const int n = XDIM * DI;
            cvt_bf16_kernel<<<(n + 255) / 256, 256, 0, stream>>>(P.xproj_w, wbuf, n);
            dim3 g(BL / 128, (XDIM + 127) / 128);
            gemm_wmma_bf16<0><<<g, 256, 0, stream>>>(uh, wbuf, xdbl, BL, XDIM, DI, nullptr, 0);
        }

        // dt input slice -> bf16
        slice_cast_kernel<<<(BL * DRANK) / 256, 256, 0, stream>>>(xdbl, dtin);

        // dt = softplus(dtin @ dt_w^T + dt_b)  (2048 x 2048, K=64)
        {
            const int n = DI * DRANK;
            cvt_bf16_kernel<<<(n + 255) / 256, 256, 0, stream>>>(P.dt_w, wbuf, n);
            dim3 g(BL / 128, DI / 128);
            gemm_wmma_bf16<1><<<g, 256, 0, stream>>>(dtin, wbuf, dtbuf, BL, DI, DRANK, P.dt_b, 0);
        }

        // selective scan + gating  (128 single-wave blocks)
        scan_kernel<<<BATCHN * (DI / 32), 32, 0, stream>>>(dtbuf, ubuf, xz, xdbl,
                                                           P.A_log, P.D, ybuf);

        // out-proj, accumulated into residual; bwd accumulates row-reversed
        {
            const int n = DM * DI;
            cvt_bf16_kernel<<<(n + 255) / 256, 256, 0, stream>>>(P.out_w, wbuf, n);
            dim3 g(BL / 128, DM / 128);
            gemm_wmma_bf16<2><<<g, 256, 0, stream>>>(ybuf, wbuf, out, BL, DM, DI,
                                                     nullptr, d ? SEQL : 0);
        }
    }
}
